// FeatureGroupBranch_7541962571817
// MI455X (gfx1250) — matrix operand order repaired and checked by reading, not compiled, not run on hardware
//
#include <hip/hip_runtime.h>
#include <hip/hip_bf16.h>

typedef __attribute__((ext_vector_type(16))) _Float16 v16h;
typedef __attribute__((ext_vector_type(8)))  _Float16 v8h;
typedef __attribute__((ext_vector_type(8)))  float    v8f;

#define NB   8
#define LL   2048
#define NF_  64
#define DM_  512
#define DS_  16
#define DC__ 4
#define DI_  1024
#define DTR_ 32
#define RROWS (NB * LL)   // 16384

__device__ __forceinline__ float siluf(float x)     { return x / (1.f + __expf(-x)); }
__device__ __forceinline__ float softplusf(float x) { return (x > 20.f) ? x : log1pf(__expf(x)); }

// ---------------- elementwise f32 -> f16 ----------------
__global__ void cvt_f32_f16_kernel(const float* __restrict__ X, _Float16* __restrict__ Y, long long n) {
  long long i = (long long)blockIdx.x * blockDim.x + threadIdx.x;
  if (i < n) Y[i] = (_Float16)X[i];
}

// ---------------- WMMA GEMM: C[m,n] = sum_k A[m,k] * W[n,k] (+bias)(+softplus) ----------------
// A: (M,K) f16 row-major (stride lda); W: (N,K) f16 row-major (stride ldw).
// One wave computes a 32x64 strip of C (2 M-tiles x 4 N-tiles = 8 WMMAs per K=32 step).
// Register layouts per CDNA5 ISA 7.12.2.
template<int EPI>
__global__ void gemm_wmma_kernel(const _Float16* __restrict__ A, int lda,
                                 const _Float16* __restrict__ W, int ldw,
                                 const float* __restrict__ bias,
                                 float* __restrict__ C, int ldc,
                                 int M, int N, int K) {
  const int tilesN = N >> 6;            // 64-wide strips
  const int tilesM = M >> 5;            // 32-tall strips
  const int wave   = (int)((blockIdx.x * blockDim.x + threadIdx.x) >> 5);
  if (wave >= tilesM * tilesN) return;
  const int tm   = wave / tilesN;
  const int tn   = wave - tm * tilesN;
  const int lane = threadIdx.x & 31;
  const int l15  = lane & 15;
  const int half = lane >> 4;   // 0: lanes 0-15, 1: lanes 16-31

  // A lane row: M = base + l15. K chunks: half==0 -> {0..7,16..23}; half==1 -> {8..15,24..31}
  const _Float16* arow0 = A + (size_t)(tm * 32 + l15) * lda;
  const _Float16* arow1 = arow0 + (size_t)16 * lda;
  // B lane column: N = tn*64 + j*16 + l15. K chunks: half==0 -> {0..7,16..23}; half==1 -> {8..15,24..31}
  const _Float16* wrow0 = W + (size_t)(tn * 64 + l15) * ldw;

  v8f acc[8];
#pragma unroll
  for (int i = 0; i < 8; ++i) acc[i] = (v8f){};

  for (int kt = 0; kt < K; kt += 32) {
    __builtin_prefetch(arow0 + kt + 32, 0, 1);
    __builtin_prefetch(arow1 + kt + 32, 0, 1);
    v16h a0, a1;
    {
      v8h x0 = *(const v8h*)(arow0 + kt + half * 8);
      v8h x1 = *(const v8h*)(arow0 + kt + 16 + half * 8);
      v8h y0 = *(const v8h*)(arow1 + kt + half * 8);
      v8h y1 = *(const v8h*)(arow1 + kt + 16 + half * 8);
#pragma unroll
      for (int i = 0; i < 8; ++i) { a0[i] = x0[i]; a0[i + 8] = x1[i]; a1[i] = y0[i]; a1[i + 8] = y1[i]; }
    }
#pragma unroll
    for (int j = 0; j < 4; ++j) {
      const _Float16* wr = wrow0 + (size_t)(j * 16) * ldw;
      v8h b0 = *(const v8h*)(wr + kt + half * 8);
      v8h b1 = *(const v8h*)(wr + kt + 16 + half * 8);
      v16h bv;
#pragma unroll
      for (int i = 0; i < 8; ++i) { bv[i] = b0[i]; bv[i + 8] = b1[i]; }
      acc[j]     = __builtin_amdgcn_wmma_f32_16x16x32_f16(false, a0, false, bv, (short)0, acc[j],     false, false);
      acc[4 + j] = __builtin_amdgcn_wmma_f32_16x16x32_f16(false, a1, false, bv, (short)0, acc[4 + j], false, false);
    }
  }
  // C layout: VGPR r -> M = base + r + half*8 ; N = col
  const int mb0 = tm * 32 + half * 8;
#pragma unroll
  for (int j = 0; j < 4; ++j) {
    const int col = tn * 64 + j * 16 + l15;
    float bb = 0.f;
    if (EPI >= 1) bb = bias[col];
#pragma unroll
    for (int r = 0; r < 8; ++r) {
      float v0 = acc[j][r] + bb;
      float v1 = acc[4 + j][r] + bb;
      if (EPI == 2) { v0 = softplusf(v0); v1 = softplusf(v1); }
      C[(size_t)(mb0 + r) * ldc + col]      = v0;
      C[(size_t)(mb0 + 16 + r) * ldc + col] = v1;
    }
  }
}

// ---------------- LayerNorm (+optional residual), emits f32 and optional f16 ----------------
__global__ void ln_res_kernel(const float* __restrict__ X, const float* __restrict__ Rr,
                              const float* __restrict__ g, const float* __restrict__ b,
                              float* __restrict__ Of32, _Float16* __restrict__ Of16, int D) {
  const int row = blockIdx.x;
  const int t   = threadIdx.x;
  const float* xr = X + (size_t)row * D;
  const float* rr = Rr ? Rr + (size_t)row * D : nullptr;
  float v[2];
  float s = 0.f;
#pragma unroll
  for (int i = 0; i < 2; ++i) {
    int c = t + i * 256;
    float x = xr[c] + (rr ? rr[c] : 0.f);
    v[i] = x; s += x;
  }
  __shared__ float red[256];
  red[t] = s; __syncthreads();
  for (int off = 128; off > 0; off >>= 1) { if (t < off) red[t] += red[t + off]; __syncthreads(); }
  const float mean = red[0] / (float)D;
  __syncthreads();
  float q = 0.f;
#pragma unroll
  for (int i = 0; i < 2; ++i) { float d0 = v[i] - mean; q += d0 * d0; }
  red[t] = q; __syncthreads();
  for (int off = 128; off > 0; off >>= 1) { if (t < off) red[t] += red[t + off]; __syncthreads(); }
  const float rstd = rsqrtf(red[0] / (float)D + 1e-5f);
#pragma unroll
  for (int i = 0; i < 2; ++i) {
    int c = t + i * 256;
    float o = (v[i] - mean) * rstd * g[c] + b[c];
    Of32[(size_t)row * D + c] = o;
    if (Of16) Of16[(size_t)row * D + c] = (_Float16)o;
  }
}

// ---------------- causal depthwise conv1d (DC=4) + SiLU ----------------
__global__ void conv_silu_kernel(const float* __restrict__ xz, const float* __restrict__ cw,
                                 const float* __restrict__ cb,
                                 float* __restrict__ u, _Float16* __restrict__ uh) {
  long long idx = (long long)blockIdx.x * blockDim.x + threadIdx.x;
  if (idx >= (long long)RROWS * DI_) return;
  const int d       = (int)(idx & (DI_ - 1));
  const long long row = idx >> 10;          // b*L + l
  const int l       = (int)(row & (LL - 1));
  float acc = cb[d];
#pragma unroll
  for (int j = 0; j < DC__; ++j) {
    int lp = l - (DC__ - 1) + j;
    if (lp >= 0) acc += cw[d * DC__ + j] * xz[(row - (DC__ - 1) + j) * (2 * DI_) + d];
  }
  float sv = siluf(acc);
  u[idx]  = sv;
  uh[idx] = (_Float16)sv;
}

// ---------------- selective scan: 1 thread per (b,d), 16 states in regs, B/C in LDS ----------------
__global__ void scan_kernel(const float* __restrict__ u, const float* __restrict__ delta,
                            const float* __restrict__ dbl, const float* __restrict__ xz,
                            const float* __restrict__ A_log, const float* __restrict__ Dv,
                            _Float16* __restrict__ yh) {
  const int b = blockIdx.x >> 2;
  const int d = ((blockIdx.x & 3) << 8) + threadIdx.x;   // 0..1023
  float A[DS_], h[DS_];
#pragma unroll
  for (int s = 0; s < DS_; ++s) { A[s] = -__expf(A_log[d * DS_ + s]); h[s] = 0.f; }
  const float Dd = Dv[d];
  __shared__ float sB[DS_], sC[DS_];
  for (int l = 0; l < LL; ++l) {
    const long long row = (long long)b * LL + l;
    if (threadIdx.x < 32) {   // cols 32..63 of dbl are B (16) then C (16)
      float vv = dbl[row * 64 + 32 + threadIdx.x];
      if (threadIdx.x < 16) sB[threadIdx.x] = vv; else sC[threadIdx.x - 16] = vv;
    }
    __syncthreads();
    const float dt = delta[row * DI_ + d];
    const float uv = u[row * DI_ + d];
    const float du = dt * uv;
    float yv = 0.f;
#pragma unroll
    for (int s = 0; s < DS_; ++s) {
      h[s] = __expf(dt * A[s]) * h[s] + du * sB[s];
      yv += h[s] * sC[s];
    }
    const float zv = xz[row * (2 * DI_) + DI_ + d];
    const float o  = (yv + uv * Dd) * siluf(zv);
    yh[row * DI_ + d] = (_Float16)o;
    __syncthreads();
  }
}

// ---------------- exact GELU -> f16 ----------------
__global__ void gelu_h_kernel(const float* __restrict__ X, _Float16* __restrict__ Y, long long n) {
  long long i = (long long)blockIdx.x * blockDim.x + threadIdx.x;
  if (i < n) {
    float x = X[i];
    Y[i] = (_Float16)(0.5f * x * (1.f + erff(x * 0.70710678118654752f)));
  }
}

// ---------------- host side ----------------
static inline size_t alignUp256(size_t x) { return (x + 255) & ~(size_t)255; }

template<int EPI>
static void launch_gemm(const _Float16* A, int lda, const _Float16* W, int ldw, const float* bias,
                        float* C, int ldc, int M, int N, int K, hipStream_t s) {
  long long waves  = (long long)(M / 32) * (N / 64);
  int       blocks = (int)((waves + 7) / 8);   // 8 waves (256 threads) per block
  gemm_wmma_kernel<EPI><<<blocks, 256, 0, s>>>(A, lda, W, ldw, bias, C, ldc, M, N, K);
}

static void launch_cvt(const float* X, _Float16* Y, long long n, hipStream_t s) {
  cvt_f32_f16_kernel<<<(int)((n + 255) / 256), 256, 0, s>>>(X, Y, n);
}

extern "C" void kernel_launch(void* const* d_in, const int* in_sizes, int n_in,
                              void* d_out, int out_size, void* d_ws, size_t ws_size,
                              hipStream_t stream) {
  const float* x         = (const float*)d_in[0];
  const float* W_in      = (const float*)d_in[1];
  const float* b_in      = (const float*)d_in[2];
  const float* ln1_g     = (const float*)d_in[3];
  const float* ln1_b     = (const float*)d_in[4];
  const float* in_proj_w = (const float*)d_in[5];
  const float* conv_w    = (const float*)d_in[6];
  const float* conv_b    = (const float*)d_in[7];
  const float* x_proj_w  = (const float*)d_in[8];
  const float* dt_proj_w = (const float*)d_in[9];
  const float* dt_proj_b = (const float*)d_in[10];
  const float* A_log     = (const float*)d_in[11];
  const float* Dv        = (const float*)d_in[12];
  const float* out_proj_w= (const float*)d_in[13];
  const float* ln2_g     = (const float*)d_in[14];
  const float* ln2_b     = (const float*)d_in[15];
  const float* W1        = (const float*)d_in[16];
  const float* b1        = (const float*)d_in[17];
  const float* W2        = (const float*)d_in[18];
  const float* b2        = (const float*)d_in[19];
  const float* ln3_g     = (const float*)d_in[20];
  const float* ln3_b     = (const float*)d_in[21];
  float* out = (float*)d_out;

  char*  ws  = (char*)d_ws;
  size_t off = 0;
  auto alloc = [&](size_t bytes) -> void* { void* p = ws + off; off = alignUp256(off + bytes); return p; };

  const size_t R = RROWS;
  _Float16* xh        = (_Float16*)alloc(R * NF_ * 2);
  _Float16* wInH      = (_Float16*)alloc((size_t)DM_ * NF_ * 2);
  _Float16* wInProjH  = (_Float16*)alloc((size_t)2 * DI_ * DM_ * 2);
  _Float16* wXProjH   = (_Float16*)alloc((size_t)64 * DI_ * 2);
  _Float16* wDtProjH  = (_Float16*)alloc((size_t)DI_ * DTR_ * 2);
  _Float16* wOutProjH = (_Float16*)alloc((size_t)DM_ * DI_ * 2);
  _Float16* w1H       = (_Float16*)alloc((size_t)2 * DM_ * DM_ * 2);
  _Float16* w2H       = (_Float16*)alloc((size_t)DM_ * 2 * DM_ * 2);
  float*    tmpDM     = (float*)   alloc(R * DM_ * 4);        // GEMM1 out / mamba out / FFN out
  float*    h1        = (float*)   alloc(R * DM_ * 4);
  _Float16* h1h       = (_Float16*)alloc(R * DM_ * 2);
  float*    xz        = (float*)   alloc(R * 2 * DI_ * 4);
  float*    u         = (float*)   alloc(R * DI_ * 4);
  _Float16* uh        = (_Float16*)alloc(R * DI_ * 2);
  float*    dbl       = (float*)   alloc(R * 64 * 4);
  _Float16* dblh      = (_Float16*)alloc(R * 64 * 2);
  float*    tmpDI     = (float*)   alloc(R * DI_ * 4);        // delta / FFN hidden
  _Float16* tmpDIh    = (_Float16*)alloc(R * DI_ * 2);        // y_f16 / gelu_f16
  float*    h2n       = (float*)   alloc(R * DM_ * 4);
  _Float16* h2nh      = (_Float16*)alloc(R * DM_ * 2);

  // 1) h = x @ W_in.T + b_in ; LN1
  launch_cvt(x, xh, (long long)R * NF_, stream);
  launch_cvt(W_in, wInH, (long long)DM_ * NF_, stream);
  launch_gemm<1>(xh, NF_, wInH, NF_, b_in, tmpDM, DM_, (int)R, DM_, NF_, stream);
  ln_res_kernel<<<(int)R, 256, 0, stream>>>(tmpDM, nullptr, ln1_g, ln1_b, h1, h1h, DM_);

  // 2) xz = h1 @ in_proj_w.T  (B,L,2*DI)
  launch_cvt(in_proj_w, wInProjH, (long long)2 * DI_ * DM_, stream);
  launch_gemm<0>(h1h, DM_, wInProjH, DM_, nullptr, xz, 2 * DI_, (int)R, 2 * DI_, DM_, stream);

  // 3) u = silu(causal depthwise conv(xc))
  conv_silu_kernel<<<(int)((R * DI_ + 255) / 256), 256, 0, stream>>>(xz, conv_w, conv_b, u, uh);

  // 4) dbl = u @ x_proj_w.T  (dt | B | C)
  launch_cvt(x_proj_w, wXProjH, (long long)64 * DI_, stream);
  launch_gemm<0>(uh, DI_, wXProjH, DI_, nullptr, dbl, 64, (int)R, 64, DI_, stream);
  launch_cvt(dbl, dblh, (long long)R * 64, stream);

  // 5) delta = softplus(dt @ dt_proj_w.T + dt_proj_b)
  launch_cvt(dt_proj_w, wDtProjH, (long long)DI_ * DTR_, stream);
  launch_gemm<2>(dblh, 64, wDtProjH, DTR_, dt_proj_b, tmpDI, DI_, (int)R, DI_, DTR_, stream);

  // 6) selective scan + D skip + silu(z) gating -> y (f16)
  scan_kernel<<<NB * (DI_ / 256), 256, 0, stream>>>(u, tmpDI, dbl, xz, A_log, Dv, tmpDIh);

  // 7) mamba_out = y @ out_proj_w.T ; h2 = LN2(h1 + mamba_out)
  launch_cvt(out_proj_w, wOutProjH, (long long)DM_ * DI_, stream);
  launch_gemm<0>(tmpDIh, DI_, wOutProjH, DI_, nullptr, tmpDM, DM_, (int)R, DM_, DI_, stream);
  ln_res_kernel<<<(int)R, 256, 0, stream>>>(tmpDM, h1, ln2_g, ln2_b, h2n, h2nh, DM_);

  // 8) FFN: gelu(h2 @ W1.T + b1) @ W2.T + b2 ; out = LN3(h2 + ffn)
  launch_cvt(W1, w1H, (long long)2 * DM_ * DM_, stream);
  launch_gemm<1>(h2nh, DM_, w1H, DM_, b1, tmpDI, 2 * DM_, (int)R, 2 * DM_, DM_, stream);
  gelu_h_kernel<<<(int)((R * DI_ + 255) / 256), 256, 0, stream>>>(tmpDI, tmpDIh, (long long)R * DI_);
  launch_cvt(W2, w2H, (long long)DM_ * 2 * DM_, stream);
  launch_gemm<1>(tmpDIh, 2 * DM_, w2H, 2 * DM_, b2, tmpDM, DM_, (int)R, DM_, 2 * DM_, stream);
  ln_res_kernel<<<(int)R, 256, 0, stream>>>(tmpDM, h2n, ln3_g, ln3_b, out, nullptr, DM_);
}
